// VectorQuantizer_66443144069606
// MI455X (gfx1250) — matrix operand order repaired and checked by reading, not compiled, not run on hardware
//
#include <hip/hip_runtime.h>
#include <hip/hip_bf16.h>

// ---------------------------------------------------------------------------
// VQ codebook lookup on MI455X (gfx1250, wave32, WMMA)
//   x:        [32768, 256] f32      codebook: [8192, 256] f32
//   out:      quantized_ste [32768*256] f32  ++  total_loss (1 f32)
//
// Cross-term GEMM via v_wmma_f32_16x16x32_bf16 with a hi/lo bf16 split of
// both operands (3 WMMAs/k-step -> ~f32 accuracy at bf16 rate).
// Codebook is pre-split/pre-swizzled into B-fragment order once, so the hot
// loop is pure WMMA + b128 loads (no conversion VALU). Argmin kept lane-local
// in the loop; one cross-lane reduction at the end.
// ---------------------------------------------------------------------------

#define DIM        256
#define NROWS      32768      // 32*1024
#define KCODES     8192
#define WAVES      4          // waves per workgroup in the argmin kernel
#define ROWS_WG    (WAVES*16) // 64 rows per workgroup
#define NTILES     (KCODES/16)   // 512 column tiles
#define KSTEPS     (DIM/32)      // 8 k-steps
#define BCHUNKS    (NTILES*KSTEPS*32)  // per-lane 64B chunks in packed B

typedef __bf16 v16bf __attribute__((ext_vector_type(16)));
typedef __bf16 v8bf  __attribute__((ext_vector_type(8)));
typedef float  v8f   __attribute__((ext_vector_type(8)));
typedef float  v4f   __attribute__((ext_vector_type(4)));

union Frag16 { v16bf v; v8bf h[2]; };

// ---------------------------------------------------------------- kernel 0 --
// Pre-split codebook -> packed hi/lo bf16 in exact B-fragment order.
// Chunk t = (ct*KSTEPS + kt)*32 + lane holds 16 hi then 16 lo bf16 (64 B).
// Fragment element e of lane L = codebook[ct*16 + (L&15)][kt*32 + (L<16?0:16) + e]
__global__ void __launch_bounds__(256)
vq_bsplit_kernel(const float* __restrict__ codebook, __bf16* __restrict__ bPacked)
{
    const int t = blockIdx.x * blockDim.x + threadIdx.x;
    if (t >= BCHUNKS) return;
    const int lane = t & 31;
    const int kt   = (t >> 5) & (KSTEPS - 1);
    const int ct   = t >> 8;
    const int code = ct * 16 + (lane & 15);
    const int kb   = kt * 32 + ((lane < 16) ? 0 : 16);

    const float* src = codebook + (size_t)code * DIM + kb;
    __bf16*      dst = bPacked + (size_t)t * 32;   // 32 bf16 = 64 bytes
#pragma unroll
    for (int e = 0; e < 16; ++e) {
        float  f = src[e];
        __bf16 h = (__bf16)f;                      // RNE hi part
        dst[e]      = h;
        dst[16 + e] = (__bf16)(f - (float)h);      // residual lo part
    }
}

// ---------------------------------------------------------------- kernel 1 --
// ||c_k||^2 for every codebook row. One wave per row: 8 f32 per lane.
__global__ void __launch_bounds__(256)
vq_cnorm_kernel(const float* __restrict__ codebook, float* __restrict__ cnorm)
{
    const int gid  = blockIdx.x * blockDim.x + threadIdx.x;
    const int wave = gid >> 5;
    const int lane = gid & 31;
    if (wave >= KCODES) return;

    const float* row = codebook + (size_t)wave * DIM + lane * 8;
    v4f a = *(const v4f*)(row);
    v4f b = *(const v4f*)(row + 4);
    float s = 0.0f;
#pragma unroll
    for (int e = 0; e < 4; ++e) { s = fmaf(a[e], a[e], s); s = fmaf(b[e], b[e], s); }
#pragma unroll
    for (int m = 1; m < 32; m <<= 1) s += __shfl_xor(s, m, 32);
    if (lane == 0) cnorm[wave] = s;
}

// ---------------------------------------------------------------- kernel 2 --
// Argmin over the codebook via WMMA.  4 waves/WG, 16 rows/wave.
// dist = ||c||^2 - 2 x.c   (||x||^2 is row-constant, irrelevant for argmin)
__global__ void __launch_bounds__(WAVES * 32)
vq_argmin_kernel(const float* __restrict__ x,
                 const __bf16* __restrict__ bPacked,
                 const float* __restrict__ cnorm,
                 int* __restrict__ bestIdx)
{
    // Per-wave hi/lo bf16 copy of its 16x256 x-block: 4 * 16KB = 64KB LDS.
    __shared__ __attribute__((aligned(16))) __bf16 aLds[WAVES * 2 * 16 * DIM];

    const int lane    = threadIdx.x & 31;
    const int wave    = threadIdx.x >> 5;
    const int rowBase = blockIdx.x * ROWS_WG + wave * 16;

    __bf16* aHi = aLds + wave * (2 * 16 * DIM);
    __bf16* aLo = aHi + 16 * DIM;

    // ---- prologue: split the wave's 16 rows of x into bf16 hi/lo in LDS ----
    const float* xblk = x + (size_t)rowBase * DIM;
#pragma unroll 4
    for (int i = lane; i < (16 * DIM) / 4; i += 32) {      // 32 iters/lane
        v4f f = *(const v4f*)(xblk + i * 4);
        const int row = (i * 4) >> 8;
        const int k   = (i * 4) & (DIM - 1);
        __bf16* ph = aHi + row * DIM + k;
        __bf16* pl = aLo + row * DIM + k;
#pragma unroll
        for (int e = 0; e < 4; ++e) {
            float  ff = f[e];
            __bf16 hv = (__bf16)ff;
            ph[e] = hv;
            pl[e] = (__bf16)(ff - (float)hv);
        }
    }

    const int col   = lane & 15;                 // N index inside a tile
    const int kSelA = (lane < 16) ? 0 : 16;      // A-fragment chunk base
    const int rowA  = lane & 15;                 // A row for this lane

    // A fragments: loop-invariant across the 512 column tiles; loaded once
    // from LDS here, the compiler keeps them register-resident (128 VGPRs).
    Frag16 ahf[KSTEPS], alf[KSTEPS];
#pragma unroll
    for (int kt = 0; kt < KSTEPS; ++kt) {
        const __bf16* pHi = aHi + rowA * DIM + kt * 32 + kSelA;
        const __bf16* pLo = aLo + rowA * DIM + kt * 32 + kSelA;
        ahf[kt].h[0] = *(const v8bf*)(pHi);
        ahf[kt].h[1] = *(const v8bf*)(pHi + 8);
        alf[kt].h[0] = *(const v8bf*)(pLo);
        alf[kt].h[1] = *(const v8bf*)(pLo + 8);
    }

    float bestD[8];
    int   bestI[8];
#pragma unroll
    for (int r = 0; r < 8; ++r) { bestD[r] = 3.4e38f; bestI[r] = 0; }

    for (int ct = 0; ct < NTILES; ++ct) {
        // this lane's packed B chunks for the tile: KSTEPS x 64 bytes
        const __bf16* bt = bPacked + ((size_t)(ct * KSTEPS) * 32 + lane) * 32;

        // prefetch next tile's chunk stream (global_prefetch_b8)
        __builtin_prefetch((const void*)(bt + (size_t)KSTEPS * 32 * 32), 0, 0);

        v8f acc = {0.f, 0.f, 0.f, 0.f, 0.f, 0.f, 0.f, 0.f};

#pragma unroll
        for (int kt = 0; kt < KSTEPS; ++kt) {
            const __bf16* p = bt + (size_t)kt * 32 * 32;   // next k-step chunk
            v16bf bhv = *(const v16bf*)(p);                // 16 hi bf16 (32 B)
            v16bf blv = *(const v16bf*)(p + 16);           // 16 lo bf16 (32 B)

            // x.c ~= xh.ch + xh.cl + xl.ch  (f32 accumulate)
            acc = __builtin_amdgcn_wmma_f32_16x16x32_bf16(false, ahf[kt].v, false, bhv,
                                                          (short)0, acc, false, false);
            acc = __builtin_amdgcn_wmma_f32_16x16x32_bf16(false, ahf[kt].v, false, blv,
                                                          (short)0, acc, false, false);
            acc = __builtin_amdgcn_wmma_f32_16x16x32_bf16(false, alf[kt].v, false, bhv,
                                                          (short)0, acc, false, false);
        }

        // lane-local running argmin (no shuffles in the hot loop).
        // Within a lane candidate indices strictly increase, so '<' keeps the
        // first occurrence, matching jnp.argmin.
        const float cn   = cnorm[ct * 16 + col];
        const int   cand = ct * 16 + col;
#pragma unroll
        for (int r = 0; r < 8; ++r) {
            float d = fmaf(-2.0f, acc[r], cn);
            if (d < bestD[r]) { bestD[r] = d; bestI[r] = cand; }
        }
    }

    // one cross-lane argmin per row: reduce the 16 lanes of each half
    // (C/D layout: VGPR r <-> rows r (lanes 0-15) and r+8 (lanes 16-31))
#pragma unroll
    for (int r = 0; r < 8; ++r) {
        float d   = bestD[r];
        int   idx = bestI[r];
#pragma unroll
        for (int m = 1; m < 16; m <<= 1) {
            float od = __shfl_xor(d, m, 32);
            int   oi = __shfl_xor(idx, m, 32);
            if (od < d || (od == d && oi < idx)) { d = od; idx = oi; }
        }
        bestD[r] = d; bestI[r] = idx;
    }

    if ((lane & 15) == 0) {
        const int rOff = (lane >> 4) * 8;
#pragma unroll
        for (int r = 0; r < 8; ++r)
            bestIdx[rowBase + rOff + r] = bestI[r];
    }
}

// ---------------------------------------------------------------- kernel 3 --
// Gather quantized rows, emit STE output x + (q - x), per-wave partial SSE.
__global__ void __launch_bounds__(256)
vq_gather_kernel(const float* __restrict__ x,
                 const float* __restrict__ codebook,
                 const int* __restrict__ bestIdx,
                 float* __restrict__ out,
                 float* __restrict__ partials)
{
    const int gid  = blockIdx.x * blockDim.x + threadIdx.x;
    const int wave = gid >> 5;
    const int lane = gid & 31;
    if (wave >= NROWS) return;

    const int    idx = bestIdx[wave];
    const float* c   = codebook + (size_t)idx * DIM + lane * 8;
    const float* xr  = x + (size_t)wave * DIM + lane * 8;
    float*       o   = out + (size_t)wave * DIM + lane * 8;

    float s = 0.0f;
#pragma unroll
    for (int h = 0; h < 2; ++h) {
        v4f q  = *(const v4f*)(c + 4 * h);
        v4f xv = *(const v4f*)(xr + 4 * h);
        v4f ov;
#pragma unroll
        for (int e = 0; e < 4; ++e) {
            float dlt = q[e] - xv[e];
            ov[e] = xv[e] + dlt;                 // reference STE op order
            s = fmaf(dlt, dlt, s);
        }
        *(v4f*)(o + 4 * h) = ov;
    }
#pragma unroll
    for (int m = 1; m < 32; m <<= 1) s += __shfl_xor(s, m, 32);
    if (lane == 0) partials[wave] = s;
}

// ---------------------------------------------------------------- kernel 4 --
// Deterministic fixed-order reduction of the 32768 partials -> total loss.
__global__ void __launch_bounds__(256)
vq_loss_kernel(const float* __restrict__ partials, float* __restrict__ lossOut)
{
    __shared__ float red[256];
    float s = 0.0f;
    for (int i = threadIdx.x; i < NROWS; i += 256) s += partials[i];
    red[threadIdx.x] = s;
    __syncthreads();
    for (int st = 128; st > 0; st >>= 1) {
        if (threadIdx.x < st) red[threadIdx.x] += red[threadIdx.x + st];
        __syncthreads();
    }
    // codebook_loss + 0.25*commitment_loss, both equal mean((q-x)^2)
    if (threadIdx.x == 0)
        *lossOut = 1.25f * red[0] / (float)(NROWS * DIM);
}

// ---------------------------------------------------------------------------
extern "C" void kernel_launch(void* const* d_in, const int* in_sizes, int n_in,
                              void* d_out, int out_size, void* d_ws, size_t ws_size,
                              hipStream_t stream)
{
    (void)in_sizes; (void)n_in; (void)out_size; (void)ws_size;

    const float* x  = (const float*)d_in[0];   // [32768, 256] f32
    const float* cb = (const float*)d_in[1];   // [8192, 256]  f32
    // d_in[2] = hidden_dim scalar (256), unused

    float* out  = (float*)d_out;               // quantized_ste, 8388608 f32
    float* loss = out + (size_t)NROWS * DIM;   // +1 f32

    char*   ws       = (char*)d_ws;
    float*  cnorm    = (float*)(ws);                          // 8192 f32  ( 32 KB)
    int*    bestIdx  = (int*)  (ws + 32 * 1024);              // 32768 i32 (128 KB)
    float*  partials = (float*)(ws + 160 * 1024);             // 32768 f32 (128 KB)
    __bf16* bPacked  = (__bf16*)(ws + 512 * 1024);            // 8 MB packed hi/lo B

    vq_bsplit_kernel<<<BCHUNKS / 256, 256, 0, stream>>>(cb, bPacked);
    vq_cnorm_kernel <<<KCODES / 8, 256, 0, stream>>>(cb, cnorm);
    vq_argmin_kernel<<<NROWS / ROWS_WG, WAVES * 32, 0, stream>>>(x, bPacked, cnorm, bestIdx);
    vq_gather_kernel<<<NROWS / 8, 256, 0, stream>>>(x, cb, bestIdx, out, partials);
    vq_loss_kernel  <<<1, 256, 0, stream>>>(partials, loss);
}
